// Propgate_time_step_75239237091552
// MI455X (gfx1250) — matrix operand order repaired and checked by reading, not compiled, not run on hardware
//
#include <hip/hip_runtime.h>
#include <hip/hip_bf16.h>
#include <math.h>

// ---------------------------------------------------------------------------
// Problem constants (from the reference)
// ---------------------------------------------------------------------------
#define NB    2048          // batch
#define KOBJ  8             // objects (sequential scan)
#define HID   256
#define GG    50            // glimpse grid
#define G2    2500          // 50*50
#define ENC   100
#define IMH   128
#define IMW   128

// padded K-dims (multiples of 64 for the 2x-unrolled double-buffered K loop)
#define DP_G1   2560        // glimpse matrix       (2500 -> 2560)
#define DP_REL  768         // [enc|zwl|pzw|zwhatl|pzwhat|hl|h_r] 718 -> 768
#define DP_TEM  3136        // [g2|zw|pzw|zwhatl|pzwhat|hl|h_r2] 3118 -> 3136
#define DP_WT   576         // [zwhatl|h_r2|h_t] 562 -> 576
#define OP_G    128         // 100 enc rows padded to 64-col group
#define OP_WT   64          // 50 zwhat rows padded to 64-col group

typedef _Float16 v16h __attribute__((ext_vector_type(16)));
typedef _Float16 v8h  __attribute__((ext_vector_type(8)));
typedef float    v8f  __attribute__((ext_vector_type(8)));
typedef _Float16 h16;

__device__ __forceinline__ float sigf(float x) { return 1.0f / (1.0f + __expf(-x)); }

// ---------------------------------------------------------------------------
// Weight conversion: f32 -> f16, concatenating [src0 | src1] per row, zero pad.
// dst is Orows x Dp (grid.x = Orows padded; rows >= Ovalid are zero-filled).
// ---------------------------------------------------------------------------
__global__ void k_convw(const float* __restrict__ s0, int w0,
                        const float* __restrict__ s1, int w1,
                        h16* __restrict__ dst, int Dp, int Ovalid) {
  int o = blockIdx.x;
  for (int c = threadIdx.x; c < Dp; c += blockDim.x) {
    float v = 0.0f;
    if (o < Ovalid) {
      if (c < w0)            v = s0[(size_t)o * w0 + c];
      else if (c < w0 + w1)  v = s1[(size_t)o * w1 + (c - w0)];
    }
    dst[(size_t)o * Dp + c] = (h16)v;
  }
}

// ---------------------------------------------------------------------------
// WMMA GEMM:  out = act(A(N x DP) @ W^T + bias)
//   A : row-major f16, stride DP (zero padded K, DP % 64 == 0, compile-time)
//   W : f16 row-major (Opad x DP)
//   Each wave computes a 16x64 output strip (4 column tiles): the A fragment
//   is reused 4x per K-chunk.  K loop unrolled 2x with ping-pong fragment
//   registers so the next chunk's 10 b128 loads fly while the current chunk's
//   4 v_wmma execute.  DP is a template parameter so every B-tile address is
//   (one base reg) + (24-bit immediate ct*16*DP*2 + k offsets) -- no pointer
//   array, no spills.  __launch_bounds__(256,1) gives the RA full VGPR room.
//   grid = (Opad/64, N/128), block = 256 (8 waves).
// Fragment layouts per CDNA5 ISA 7.12.2 (16-bit A 16x32; B 32x16):
//   A lane l: row l%16, halves at k0+8*(l/16)..+7 and k0+16+8*(l/16)..+7
//   B lane l: W row (= output col) n, halves at k0+8*(l/16)..+7 and k0+16+8*(l/16)..+7
// ---------------------------------------------------------------------------
template <int DP>
__global__ __launch_bounds__(256, 1)
void k_wmma_gemm(const h16* __restrict__ A,
                 const h16* __restrict__ W,
                 const float* __restrict__ bias,
                 float* __restrict__ dstF, int dstFStride, int dstFOff,
                 h16*   __restrict__ dstH, int dstHStride, int dstHOff,
                 int O, int relu) {
  const int wave = threadIdx.x >> 5;
  const int lane = threadIdx.x & 31;
  const int rowTile = blockIdx.y * 8 + wave;   // 16 rows of A
  const int colBase = blockIdx.x * 64;         // 64 output columns (4 tiles)
  const int mr = lane & 15;
  const int hi = lane >> 4;

  const h16* Ar = A + (size_t)(rowTile * 16 + mr) * DP;
  const h16* W0 = W + (size_t)(colBase + mr) * DP;   // single base; tiles via imm

  auto loadA = [&](int k0, v16h& af) {
    v8h a0 = *(const v8h*)(Ar + k0 + 8 * hi);        // K = k0+8hi .. +7
    v8h a1 = *(const v8h*)(Ar + k0 + 16 + 8 * hi);   // K = k0+16+8hi .. +7
#pragma unroll
    for (int i = 0; i < 8; ++i) { af[i] = a0[i]; af[i + 8] = a1[i]; }
  };
  auto loadB4 = [&](int k0, v16h* bf) {
#pragma unroll
    for (int ct = 0; ct < 4; ++ct) {                 // ct*16*DP is a constant
      v8h b0 = *(const v8h*)(W0 + ct * 16 * DP + k0 + 8 * hi);
      v8h b1 = *(const v8h*)(W0 + ct * 16 * DP + k0 + 16 + 8 * hi);
#pragma unroll
      for (int i = 0; i < 8; ++i) { bf[ct][i] = b0[i]; bf[ct][i + 8] = b1[i]; }
    }
  };

  v8f acc[4];
#pragma unroll
  for (int ct = 0; ct < 4; ++ct) acc[ct] = (v8f){0.f,0.f,0.f,0.f,0.f,0.f,0.f,0.f};

  v16h af0, af1;
  v16h bf0[4], bf1[4];

  // prologue: preload chunk 0
  loadA(0, af0);
  loadB4(0, bf0);

  for (int k0 = 0; k0 < DP; k0 += 64) {
    // issue loads for chunk k0+32 (always exists: DP % 64 == 0)
    loadA(k0 + 32, af1);
    loadB4(k0 + 32, bf1);
    // prefetch the A stream ahead (global_prefetch_b8)
    if (k0 + 256 < DP) __builtin_prefetch(Ar + k0 + 256, 0, 1);
    // compute chunk k0 while the above loads are in flight
#pragma unroll
    for (int ct = 0; ct < 4; ++ct)
      acc[ct] = __builtin_amdgcn_wmma_f32_16x16x32_f16(
          false, af0, false, bf0[ct], (short)0, acc[ct], false, false);

    // issue loads for chunk k0+64 (skip on the final iteration)
    if (k0 + 64 < DP) {
      loadA(k0 + 64, af0);
      loadB4(k0 + 64, bf0);
    }
    // compute chunk k0+32
#pragma unroll
    for (int ct = 0; ct < 4; ++ct)
      acc[ct] = __builtin_amdgcn_wmma_f32_16x16x32_f16(
          false, af1, false, bf1[ct], (short)0, acc[ct], false, false);
  }

  // stores (after all WMMAs -> EXEC all-1s throughout the K loop)
#pragma unroll
  for (int ct = 0; ct < 4; ++ct) {
    const int nc = colBase + ct * 16 + mr;           // output column of this lane
    if (nc >= O) continue;
    const float bv = bias ? bias[nc] : 0.0f;
#pragma unroll
    for (int r = 0; r < 8; ++r) {                    // D vgpr r -> row r + 8*hi
      const int mo = rowTile * 16 + r + 8 * hi;
      float v = acc[ct][r] + bv;
      if (relu) v = fmaxf(v, 0.0f);
      if (dstF) dstF[(size_t)mo * dstFStride + dstFOff + nc] = v;
      if (dstH) dstH[(size_t)mo * dstHStride + dstHOff + nc] = (h16)v;
    }
  }
}

// ---------------------------------------------------------------------------
// Bilinear STN glimpse: one block per batch element, threads stride over 2500
// grid points. Writes f16 directly into the padded GEMM activation matrix.
// ---------------------------------------------------------------------------
__global__ void k_glimpse(const float* __restrict__ img,
                          const float* __restrict__ z, int zstride, int zoff,
                          h16* __restrict__ dst, int dstStride) {
  const int n = blockIdx.x;
  const float s  = z[(size_t)n * zstride + zoff + 0];
  const float tx = z[(size_t)n * zstride + zoff + 1];
  const float ty = z[(size_t)n * zstride + zoff + 2];
  const float* im = img + (size_t)n * IMH * IMW;
  for (int p = threadIdx.x; p < G2; p += blockDim.x) {
    const int i = p / GG, j = p % GG;
    const float gy = -1.0f + 2.0f * (float)i / (float)(GG - 1);
    const float gx = -1.0f + 2.0f * (float)j / (float)(GG - 1);
    const float px = (s * gx + tx + 1.0f) * 0.5f * (float)(IMW - 1);
    const float py = (s * gy + ty + 1.0f) * 0.5f * (float)(IMH - 1);
    const float x0 = floorf(px), y0 = floorf(py);
    const float wx = px - x0, wy = py - y0;
    const int x0i = (int)x0, y0i = (int)y0;
    float acc = 0.0f;
#pragma unroll
    for (int dy = 0; dy < 2; ++dy) {
#pragma unroll
      for (int dx = 0; dx < 2; ++dx) {
        const int yi = y0i + dy, xi = x0i + dx;
        const bool valid = (yi >= 0) & (yi < IMH) & (xi >= 0) & (xi < IMW);
        const int yc = yi < 0 ? 0 : (yi > IMH - 1 ? IMH - 1 : yi);
        const int xc = xi < 0 ? 0 : (xi > IMW - 1 ? IMW - 1 : xi);
        const float v = valid ? im[yc * IMW + xc] : 0.0f;
        acc += (dy ? wy : (1.0f - wy)) * (dx ? wx : (1.0f - wx)) * v;
      }
    }
    dst[(size_t)n * dstStride + p] = (h16)acc;
  }
}

// ---------------------------------------------------------------------------
// Step head: zwb = relu(hl @ W_loca^T + b) + zwl, and scatter this step's
// f16 inputs into XREL / XTEM / XWT (enc, h_r, h_t slots are filled later).
// One block per batch element, 256 threads.
// ---------------------------------------------------------------------------
__global__ void k_zwb(const float* __restrict__ hidden_last,
                      const float* __restrict__ z_where_last,
                      const float* __restrict__ z_what_last,
                      const float* __restrict__ Wloca, const float* __restrict__ bloca,
                      const float* __restrict__ prev_zw,     // d_out z_where (step k-1)
                      const float* __restrict__ prev_zwhat,  // d_out z_what  (step k-1)
                      float* __restrict__ zwb,
                      h16* __restrict__ XREL, h16* __restrict__ XTEM,
                      h16* __restrict__ XWT, int kk) {
  const int n = blockIdx.x, t = threadIdx.x;
  __shared__ float sh[HID];
  const size_t ib = (size_t)n * KOBJ + kk;
  const float hl = hidden_last[ib * HID + t];
  sh[t] = hl;
  XREL[(size_t)n * DP_REL + 206 + t] = (h16)hl;
  XTEM[(size_t)n * DP_TEM + 2606 + t] = (h16)hl;
  if (t < 50) {
    const float w = z_what_last[ib * 50 + t];
    XREL[(size_t)n * DP_REL + 106 + t] = (h16)w;
    XTEM[(size_t)n * DP_TEM + 2506 + t] = (h16)w;
    XWT[(size_t)n * DP_WT + t] = (h16)w;
    const float pw = (kk == 0) ? 0.0f : prev_zwhat[((size_t)n * KOBJ + kk - 1) * 50 + t];
    XREL[(size_t)n * DP_REL + 156 + t] = (h16)pw;
    XTEM[(size_t)n * DP_TEM + 2556 + t] = (h16)pw;
  }
  __syncthreads();
  if (t < 3) {
    const float zwl = z_where_last[ib * 3 + t];
    XREL[(size_t)n * DP_REL + 100 + t] = (h16)zwl;
    const float pz = (kk == 0) ? 0.0f : prev_zw[((size_t)n * KOBJ + kk - 1) * 3 + t];
    XREL[(size_t)n * DP_REL + 103 + t] = (h16)pz;
    XTEM[(size_t)n * DP_TEM + 2503 + t] = (h16)pz;
    float acc = bloca[t];
    for (int j = 0; j < HID; ++j) acc += sh[j] * Wloca[t * HID + j];
    zwb[(size_t)n * 3 + t] = fmaxf(acc, 0.0f) + zwl;
  }
}

// ---------------------------------------------------------------------------
// Relational LSTM pointwise + zw head.  Gates come from the WMMA GEMM (raw,
// no bias).  Updates carry (h_r, c_r), scatters h_r2 (f16) into the next
// GEMM matrices, computes zw, writes z_where output.
// ---------------------------------------------------------------------------
__global__ void k_lstm_rel(const float* __restrict__ Gm,
                           const float* __restrict__ bih, const float* __restrict__ bhh,
                           float* __restrict__ h_r, float* __restrict__ c_r,
                           const float* __restrict__ z_where_last,
                           const float* __restrict__ Wm_wh, const float* __restrict__ bm_wh,
                           float* __restrict__ out_zwhere,
                           h16* __restrict__ XREL, h16* __restrict__ XTEM,
                           h16* __restrict__ XWT, int kk) {
  const int n = blockIdx.x, j = threadIdx.x;
  __shared__ float sh[HID];
  const size_t gb = (size_t)n * 1024;
  const float gi = Gm[gb + j]        + bih[j]        + bhh[j];
  const float gf = Gm[gb + 256 + j]  + bih[256 + j]  + bhh[256 + j];
  const float gg = Gm[gb + 512 + j]  + bih[512 + j]  + bhh[512 + j];
  const float go = Gm[gb + 768 + j]  + bih[768 + j]  + bhh[768 + j];
  const float c2 = sigf(gf) * c_r[(size_t)n * HID + j] + sigf(gi) * tanhf(gg);
  const float h2 = sigf(go) * tanhf(c2);
  c_r[(size_t)n * HID + j] = c2;
  h_r[(size_t)n * HID + j] = h2;
  XREL[(size_t)n * DP_REL + 462 + j] = (h16)h2;
  XTEM[(size_t)n * DP_TEM + 2862 + j] = (h16)h2;
  XWT[(size_t)n * DP_WT + 50 + j] = (h16)h2;
  sh[j] = h2;
  __syncthreads();
  if (j < 3) {
    float acc = bm_wh[j];                       // zw = [zwl|h_r2] @ Wm_wh^T + b
    for (int q = 0; q < 3; ++q)
      acc += z_where_last[((size_t)n * KOBJ + kk) * 3 + q] * Wm_wh[j * 259 + q];
    for (int q = 0; q < HID; ++q)
      acc += sh[q] * Wm_wh[j * 259 + 3 + q];
    out_zwhere[((size_t)n * KOBJ + kk) * 3 + j] = acc;
    XTEM[(size_t)n * DP_TEM + 2500 + j] = (h16)acc;
  }
}

// ---------------------------------------------------------------------------
// Temporal LSTM pointwise. c input = c_r2 (carry, NOT modified). h_t goes to
// the h_temp output and to the zwhat GEMM matrix.
// ---------------------------------------------------------------------------
__global__ void k_lstm_tem(const float* __restrict__ Gm,
                           const float* __restrict__ bih, const float* __restrict__ bhh,
                           const float* __restrict__ c_r,
                           float* __restrict__ out_ht, h16* __restrict__ XWT, int kk) {
  const int n = blockIdx.x, j = threadIdx.x;
  const size_t gb = (size_t)n * 1024;
  const float gi = Gm[gb + j]        + bih[j]        + bhh[j];
  const float gf = Gm[gb + 256 + j]  + bih[256 + j]  + bhh[256 + j];
  const float gg = Gm[gb + 512 + j]  + bih[512 + j]  + bhh[512 + j];
  const float go = Gm[gb + 768 + j]  + bih[768 + j]  + bhh[768 + j];
  const float ct = sigf(gf) * c_r[(size_t)n * HID + j] + sigf(gi) * tanhf(gg);
  const float ht = sigf(go) * tanhf(ct);
  out_ht[((size_t)n * KOBJ + kk) * HID + j] = ht;
  XWT[(size_t)n * DP_WT + 306 + j] = (h16)ht;
}

// ---------------------------------------------------------------------------
// Presence head: p = sigmoid(zp_cal.Wm+bm)*sigmoid(zp_cal.Ws+bs)*zpl
// zp_cal = [zwhat(50) | zw(3) | h_r2(256) | h_t(256)].  One wave per element.
// ---------------------------------------------------------------------------
__global__ void k_pres(const float* __restrict__ out_zwhat,
                       const float* __restrict__ out_zwhere,
                       const float* __restrict__ h_r,
                       const float* __restrict__ out_ht,
                       const float* __restrict__ Wm, const float* __restrict__ bm,
                       const float* __restrict__ Ws, const float* __restrict__ bs,
                       const float* __restrict__ z_pres_last,
                       float* __restrict__ out_zpres, int kk) {
  const int wave = threadIdx.x >> 5, lane = threadIdx.x & 31;
  const int n = blockIdx.x * 8 + wave;
  const size_t ib = (size_t)n * KOBJ + kk;
  float am = 0.0f, as_ = 0.0f;
  for (int idx = lane; idx < 565; idx += 32) {
    float v;
    if (idx < 50)       v = out_zwhat[ib * 50 + idx];
    else if (idx < 53)  v = out_zwhere[ib * 3 + (idx - 50)];
    else if (idx < 309) v = h_r[(size_t)n * HID + (idx - 53)];
    else                v = out_ht[ib * HID + (idx - 309)];
    am  += v * Wm[idx];
    as_ += v * Ws[idx];
  }
#pragma unroll
  for (int off = 16; off > 0; off >>= 1) {
    am  += __shfl_down(am, off, 32);
    as_ += __shfl_down(as_, off, 32);
  }
  if (lane == 0)
    out_zpres[ib] = sigf(am + bm[0]) * sigf(as_ + bs[0]) * z_pres_last[ib];
}

// ---------------------------------------------------------------------------
// Host-side launch
// ---------------------------------------------------------------------------
extern "C" void kernel_launch(void* const* d_in, const int* in_sizes, int n_in,
                              void* d_out, int out_size, void* d_ws, size_t ws_size,
                              hipStream_t stream) {
  const float* img      = (const float*)d_in[0];
  const float* zwhatL   = (const float*)d_in[1];
  const float* zwhereL  = (const float*)d_in[2];
  const float* zpresL   = (const float*)d_in[3];
  const float* hiddenL  = (const float*)d_in[4];
  const float* W_loca   = (const float*)d_in[5];
  const float* b_loca   = (const float*)d_in[6];
  const float* Wg       = (const float*)d_in[7];
  const float* bg       = (const float*)d_in[8];
  const float* Wih_rel  = (const float*)d_in[9];
  const float* Whh_rel  = (const float*)d_in[10];
  const float* bih_rel  = (const float*)d_in[11];
  const float* bhh_rel  = (const float*)d_in[12];
  const float* Wih_tem  = (const float*)d_in[13];
  const float* Whh_tem  = (const float*)d_in[14];
  const float* bih_tem  = (const float*)d_in[15];
  const float* bhh_tem  = (const float*)d_in[16];
  const float* Wm_wh    = (const float*)d_in[17];
  const float* bm_wh    = (const float*)d_in[18];
  const float* Wm_wt    = (const float*)d_in[21];
  const float* bm_wt    = (const float*)d_in[22];
  const float* Wm_pr    = (const float*)d_in[25];
  const float* bm_pr    = (const float*)d_in[26];
  const float* Ws_pr    = (const float*)d_in[27];
  const float* bs_pr    = (const float*)d_in[28];

  // outputs (tuple concatenated flat, f32)
  float* out_zwhat  = (float*)d_out;                       // N*K*50
  float* out_zwhere = out_zwhat  + (size_t)NB * KOBJ * 50; // N*K*3
  float* out_zpres  = out_zwhere + (size_t)NB * KOBJ * 3;  // N*K*1
  float* out_ht     = out_zpres  + (size_t)NB * KOBJ * 1;  // N*K*256

  // workspace carve-up (all region sizes are multiples of 256B)
  char* ws = (char*)d_ws;
  size_t off = 0;
  h16* XG1  = (h16*)(ws + off); off += (size_t)NB * DP_G1 * 2;
  h16* XREL = (h16*)(ws + off); off += (size_t)NB * DP_REL * 2;
  h16* XTEM = (h16*)(ws + off); off += (size_t)NB * DP_TEM * 2;
  h16* XWT  = (h16*)(ws + off); off += (size_t)NB * DP_WT * 2;
  const size_t x_bytes = off;                                // zeroed region
  h16* WGh   = (h16*)(ws + off); off += (size_t)OP_G  * DP_G1  * 2;
  h16* WRELh = (h16*)(ws + off); off += (size_t)1024  * DP_REL * 2;
  h16* WTEMh = (h16*)(ws + off); off += (size_t)1024  * DP_TEM * 2;
  h16* WWTh  = (h16*)(ws + off); off += (size_t)OP_WT * DP_WT  * 2;
  float* GBUF = (float*)(ws + off); off += (size_t)NB * 1024 * 4;  // shared rel/tem gates
  float* h_r  = (float*)(ws + off); off += (size_t)NB * HID * 4;
  float* c_r  = (float*)(ws + off); off += (size_t)NB * HID * 4;
  float* zwb  = (float*)(ws + off); off += (size_t)NB * 3 * 4;

  // zero activation matrices (pad cols) and the LSTM carry
  hipMemsetAsync(XG1, 0, x_bytes, stream);
  hipMemsetAsync(h_r, 0, (size_t)NB * HID * 4 * 2, stream);  // h_r then c_r contiguous

  // weight conversion (fused [Wih|Whh] for the LSTMs)
  k_convw<<<OP_G, 256, 0, stream>>>(Wg, G2, nullptr, 0, WGh, DP_G1, ENC);
  k_convw<<<1024, 256, 0, stream>>>(Wih_rel, 462, Whh_rel, HID, WRELh, DP_REL, 1024);
  k_convw<<<1024, 256, 0, stream>>>(Wih_tem, 2862, Whh_tem, HID, WTEMh, DP_TEM, 1024);
  k_convw<<<OP_WT, 256, 0, stream>>>(Wm_wt, 562, nullptr, 0, WWTh, DP_WT, 50);

  const dim3 blk(256);
  for (int kk = 0; kk < KOBJ; ++kk) {
    // zwb + input scatter
    k_zwb<<<NB, blk, 0, stream>>>(hiddenL, zwhereL, zwhatL, W_loca, b_loca,
                                  out_zwhere, out_zwhat, zwb, XREL, XTEM, XWT, kk);
    // glimpse 1 (zwb)
    k_glimpse<<<NB, blk, 0, stream>>>(img, zwb, 3, 0, XG1, DP_G1);
    // enc = relu(glimpse @ Wg^T + bg) -> XREL[:,0:100] (f16)
    k_wmma_gemm<DP_G1><<<dim3(OP_G / 64, NB / 128), blk, 0, stream>>>(
        XG1, WGh, bg, nullptr, 0, 0, XREL, DP_REL, 0, ENC, 1);
    // relational LSTM gates (raw)
    k_wmma_gemm<DP_REL><<<dim3(1024 / 64, NB / 128), blk, 0, stream>>>(
        XREL, WRELh, nullptr, GBUF, 1024, 0, nullptr, 0, 0, 1024, 0);
    k_lstm_rel<<<NB, blk, 0, stream>>>(GBUF, bih_rel, bhh_rel, h_r, c_r,
                                       zwhereL, Wm_wh, bm_wh, out_zwhere,
                                       XREL, XTEM, XWT, kk);
    // glimpse 2 (zw, read from the z_where output we just wrote)
    k_glimpse<<<NB, blk, 0, stream>>>(img, out_zwhere, KOBJ * 3, kk * 3, XTEM, DP_TEM);
    // temporal LSTM gates (raw)
    k_wmma_gemm<DP_TEM><<<dim3(1024 / 64, NB / 128), blk, 0, stream>>>(
        XTEM, WTEMh, nullptr, GBUF, 1024, 0, nullptr, 0, 0, 1024, 0);
    k_lstm_tem<<<NB, blk, 0, stream>>>(GBUF, bih_tem, bhh_tem, c_r, out_ht, XWT, kk);
    // zwhat = [zwhatl|h_r2|h_t] @ Wm_wt^T + b  -> d_out (next step reads pzwhat here)
    k_wmma_gemm<DP_WT><<<dim3(OP_WT / 64, NB / 128), blk, 0, stream>>>(
        XWT, WWTh, bm_wt, out_zwhat, KOBJ * 50, kk * 50, nullptr, 0, 0, 50, 0);
    // presence
    k_pres<<<NB / 8, blk, 0, stream>>>(out_zwhat, out_zwhere, h_r, out_ht,
                                       Wm_pr, bm_pr, Ws_pr, bs_pr, zpresL,
                                       out_zpres, kk);
  }
}
